// GPET_11287174053992
// MI455X (gfx1250) — matrix operand order repaired and checked by reading, not compiled, not run on hardware
//
#include <hip/hip_runtime.h>
#include <hip/hip_bf16.h>

// ---------------------------------------------------------------------------
// CDNA5 (gfx1250): transformer encoder + MLP + prototype head.
//  - all GEMM operands pre-converted to fp16 once (half the bytes, no per-use
//    v_cvt), v_wmma_f32_16x16x32_f16 with fp32 accumulate
//  - 64x64 register-blocked GEMM (16 WMMAs / K-step, 4x fragment reuse)
//  - flash attention: 32 queries/wave, V staged to LDS with
//    global_load_async_to_lds_b128 (ASYNCcnt) overlapping the score compute
// ---------------------------------------------------------------------------

typedef __attribute__((ext_vector_type(16))) _Float16 v16h;
typedef __attribute__((ext_vector_type(8)))  _Float16 v8h;
typedef __attribute__((ext_vector_type(8)))  float    v8f;

#define B_SZ 2048
#define D_SZ 512
#define QKV_LD (3 * D_SZ)   // 1536
#define HD_SZ 64
#define NH 8
#define Q_SZ 128
#define P_SZ 40
#define EPSF 1e-5f

// Build a 16-half fragment from two 16B-aligned runs of 8 halves.
__device__ __forceinline__ v16h load16x(const _Float16* __restrict__ p0,
                                        const _Float16* __restrict__ p1) {
  v16h r;
  ((v8h*)&r)[0] = *(const v8h*)p0;
  ((v8h*)&r)[1] = *(const v8h*)p1;
  return r;
}

// ---------------------------------------------------------------------------
// fp32 -> fp16 bulk convert (weights / x), 4 elems per thread.
// ---------------------------------------------------------------------------
__global__ __launch_bounds__(256)
void cvt_f32_f16_kernel(const float* __restrict__ src, _Float16* __restrict__ dst,
                        int n) {
  const int i = (blockIdx.x * 256 + threadIdx.x) * 4;
  if (i + 3 < n) {
    const float4 f = *(const float4*)(src + i);
    dst[i + 0] = (_Float16)f.x; dst[i + 1] = (_Float16)f.y;
    dst[i + 2] = (_Float16)f.z; dst[i + 3] = (_Float16)f.w;
  }
}

// ---------------------------------------------------------------------------
// out = act(A[M,K](f16) @ W[N,K]^T(f16) + bias), one wave per 64x64 tile.
// A-fragment: lane l holds row (l&15); k run [kb+16*(l>>4)..+16).
// B-fragment: lane l holds col (l&15); k run [kb+16*(l>>4)..+16).
// Writes fp32 (outf) and/or fp16 (outh).
// ---------------------------------------------------------------------------
__global__ __launch_bounds__(32)
void gemm_bias_kernel(const _Float16* __restrict__ A, const _Float16* __restrict__ W,
                      const float* __restrict__ bias, float* __restrict__ outf,
                      _Float16* __restrict__ outh, int N, int K, int relu) {
  const int lane = threadIdx.x;
  const int hi = lane >> 4;
  const int lo = lane & 15;
  const int nb = blockIdx.x * 64;
  const int mb = blockIdx.y * 64;

  const _Float16* arow[4];
  const _Float16* wrow[4];
#pragma unroll
  for (int i = 0; i < 4; ++i) {
    arow[i] = A + (size_t)(mb + i * 16 + lo) * K;
    wrow[i] = W + (size_t)(nb + i * 16 + lo) * K;
  }

  v8f acc[4][4];
#pragma unroll
  for (int i = 0; i < 4; ++i)
#pragma unroll
    for (int j = 0; j < 4; ++j) acc[i][j] = v8f{};

  for (int kb = 0; kb < K; kb += 32) {
    if (kb + 256 < K) {   // global_prefetch_b8: pull a future K block toward cache
      __builtin_prefetch(arow[0] + kb + 256, 0, 1);
      __builtin_prefetch(wrow[0] + kb + 256, 0, 1);
    }
    v16h af[4], bf[4];
#pragma unroll
    for (int i = 0; i < 4; ++i)
      af[i] = load16x(arow[i] + kb + hi * 16, arow[i] + kb + hi * 16 + 8);
#pragma unroll
    for (int j = 0; j < 4; ++j)
      bf[j] = load16x(wrow[j] + kb + hi * 16, wrow[j] + kb + hi * 16 + 8);
#pragma unroll
    for (int i = 0; i < 4; ++i)
#pragma unroll
      for (int j = 0; j < 4; ++j)
        acc[i][j] = __builtin_amdgcn_wmma_f32_16x16x32_f16(
            false, af[i], false, bf[j], (short)0, acc[i][j], false, false);
  }

  // C layout: vgpr r, lane l -> row = r + 8*(l>>4), col = l&15
#pragma unroll
  for (int j = 0; j < 4; ++j) {
    const int col = nb + j * 16 + lo;
    const float bv = bias[col];
#pragma unroll
    for (int i = 0; i < 4; ++i) {
#pragma unroll
      for (int r = 0; r < 8; ++r) {
        const int row = mb + i * 16 + r + hi * 8;
        float v = acc[i][j][r] + bv;
        if (relu) v = fmaxf(v, 0.0f);
        if (outf) outf[(size_t)row * N + col] = v;
        if (outh) outh[(size_t)row * N + col] = (_Float16)v;
      }
    }
  }
}

// ---------------------------------------------------------------------------
// Flash attention on fp16 qkv, one wave per (32-query tile, head).
// K fragments reused across the two 16-query subtiles; V tile copied to LDS
// with global_load_async_to_lds_b128 while scores are computed.
// qkv rows: [q | k | v], head h owns cols h*64..+63. Output ctx is fp16.
// ---------------------------------------------------------------------------
__global__ __launch_bounds__(32)
void flash_attn_kernel(const _Float16* __restrict__ qkv, _Float16* __restrict__ ctx) {
  const int lane = threadIdx.x;
  const int hi = lane >> 4;
  const int lo = lane & 15;
  const int qt = blockIdx.x;   // 0..63, 32 query rows each
  const int h  = blockIdx.y;   // 0..7

  __shared__ _Float16 lds_p[2][16 * 32];
  __shared__ _Float16 lds_v[32 * 64];

  // Q fragments: two 16-row subtiles x two K=32 hd chunks
  v16h qf[2][2];
#pragma unroll
  for (int qs = 0; qs < 2; ++qs) {
    const _Float16* qrow = qkv + (size_t)(qt * 32 + qs * 16 + lo) * QKV_LD + h * HD_SZ;
#pragma unroll
    for (int c = 0; c < 2; ++c)
      qf[qs][c] = load16x(qrow + c * 32 + hi * 8, qrow + c * 32 + 16 + hi * 8);
  }

  v8f o[2][4];
  float mrun[2][8], lrun[2][8];
#pragma unroll
  for (int qs = 0; qs < 2; ++qs) {
#pragma unroll
    for (int c = 0; c < 4; ++c) o[qs][c] = v8f{};
#pragma unroll
    for (int r = 0; r < 8; ++r) { mrun[qs][r] = -1e30f; lrun[qs][r] = 0.0f; }
  }
  const float scale = 0.125f;  // 1/sqrt(64)
  const unsigned ldsa = (unsigned)(uintptr_t)&lds_v[lane * 64];

  for (int jj = 0; jj < B_SZ / 32; ++jj) {
    // ---- async-stage V tile (32 rows x 64 hd, fp16 = 128B/row), one row/lane.
    // Copies proceed in the background (ASYNCcnt) while scores are computed.
    {
      const _Float16* vrow = qkv + (size_t)(jj * 32 + lane) * QKV_LD
                           + 2 * D_SZ + h * HD_SZ;
#pragma unroll
      for (int t = 0; t < 8; ++t)
        asm volatile("global_load_async_to_lds_b128 %0, %1, off offset:%2"
                     :: "v"(ldsa), "v"(vrow), "i"(t * 16) : "memory");
    }

    // ---- K fragments: 2 key halves x 2 hd chunks (shared by both subtiles)
    v16h kf[2][2];
#pragma unroll
    for (int hh = 0; hh < 2; ++hh) {
      const _Float16* krow = qkv + (size_t)(jj * 32 + hh * 16 + lo) * QKV_LD
                           + D_SZ + h * HD_SZ;
#pragma unroll
      for (int c = 0; c < 2; ++c)
        kf[hh][c] = load16x(krow + c * 32 + hi * 8, krow + c * 32 + 16 + hi * 8);
    }

    // ---- scores + online softmax per 16-query subtile
#pragma unroll
    for (int qs = 0; qs < 2; ++qs) {
      v8f s[2];
#pragma unroll
      for (int hh = 0; hh < 2; ++hh) {
        v8f sc = {};
#pragma unroll
        for (int c = 0; c < 2; ++c)
          sc = __builtin_amdgcn_wmma_f32_16x16x32_f16(
              false, qf[qs][c], false, kf[hh][c], (short)0, sc, false, false);
        s[hh] = sc;
      }
#pragma unroll
      for (int r = 0; r < 8; ++r) {
        float a0 = s[0][r] * scale;
        float a1 = s[1][r] * scale;
        float v = fmaxf(a0, a1);
        v = fmaxf(v, __shfl_xor(v, 1, 16));
        v = fmaxf(v, __shfl_xor(v, 2, 16));
        v = fmaxf(v, __shfl_xor(v, 4, 16));
        v = fmaxf(v, __shfl_xor(v, 8, 16));
        const float mnew  = fmaxf(mrun[qs][r], v);
        const float alpha = __expf(mrun[qs][r] - mnew);
        a0 = __expf(a0 - mnew);
        a1 = __expf(a1 - mnew);
        float rs = a0 + a1;
        rs += __shfl_xor(rs, 1, 16);
        rs += __shfl_xor(rs, 2, 16);
        rs += __shfl_xor(rs, 4, 16);
        rs += __shfl_xor(rs, 8, 16);
        lrun[qs][r] = lrun[qs][r] * alpha + rs;
        mrun[qs][r] = mnew;
#pragma unroll
        for (int c = 0; c < 4; ++c) o[qs][c][r] *= alpha;
        // P: C-layout -> LDS (row-major 16x32 fp16) for A-layout reload
        lds_p[qs][(r + hi * 8) * 32 + lo]      = (_Float16)a0;
        lds_p[qs][(r + hi * 8) * 32 + 16 + lo] = (_Float16)a1;
      }
    }

    asm volatile("s_wait_dscnt 0" ::: "memory");     // P bounce visible

    // ---- reload P as A fragments
    v16h pf[2];
#pragma unroll
    for (int qs = 0; qs < 2; ++qs) {
#pragma unroll
      for (int idx = 0; idx < 16; ++idx) {
        const int k = ((idx & 8) << 1) + hi * 8 + (idx & 7);  // A-layout k index
        pf[qs][idx] = lds_p[qs][lo * 32 + k];
      }
    }

    asm volatile("s_wait_asynccnt 0" ::: "memory");  // V tile landed in LDS

    // ---- O += P @ V : V B-fragments gathered from LDS, reused by both subtiles
#pragma unroll
    for (int c = 0; c < 4; ++c) {
      v16h vf;  // B element idx: key = ((idx&8)<<1) + 8*hi + (idx&7), lds_v[key*64 + c*16 + n]
#pragma unroll
      for (int idx = 0; idx < 16; ++idx)
        vf[idx] = lds_v[(((idx & 8) << 1) + hi * 8 + (idx & 7)) * 64 + c * 16 + lo];
#pragma unroll
      for (int qs = 0; qs < 2; ++qs)
        o[qs][c] = __builtin_amdgcn_wmma_f32_16x16x32_f16(
            false, pf[qs], false, vf, (short)0, o[qs][c], false, false);
    }
  }

  // ---- normalize and write fp16 ctx
#pragma unroll
  for (int qs = 0; qs < 2; ++qs) {
#pragma unroll
    for (int r = 0; r < 8; ++r) {
      const float inv = 1.0f / lrun[qs][r];
      const int row = qt * 32 + qs * 16 + r + hi * 8;
      _Float16* crow = ctx + (size_t)row * D_SZ + h * HD_SZ + lo;
      crow[0]  = (_Float16)(o[qs][0][r] * inv);
      crow[16] = (_Float16)(o[qs][1][r] * inv);
      crow[32] = (_Float16)(o[qs][2][r] * inv);
      crow[48] = (_Float16)(o[qs][3][r] * inv);
    }
  }
}

// ---------------------------------------------------------------------------
// Row-wise instance norm (affine=False); fp32 out + optional fp16 copy.
// ---------------------------------------------------------------------------
__global__ __launch_bounds__(32)
void inorm_kernel(const float* __restrict__ in, float* __restrict__ out,
                  _Float16* __restrict__ outh, int N) {
  const int row = blockIdx.x, lane = threadIdx.x;
  const float* p = in + (size_t)row * N;
  float s = 0.0f, s2 = 0.0f;
  for (int i = lane; i < N; i += 32) { const float v = p[i]; s += v; s2 += v * v; }
#pragma unroll
  for (int off = 16; off > 0; off >>= 1) {
    s  += __shfl_xor(s,  off, 32);
    s2 += __shfl_xor(s2, off, 32);
  }
  const float mean = s / (float)N;
  const float var  = s2 / (float)N - mean * mean;
  const float rs   = rsqrtf(var + EPSF);
  float* q = out + (size_t)row * N;
  for (int i = lane; i < N; i += 32) {
    const float v = (p[i] - mean) * rs;
    q[i] = v;
    if (outh) outh[(size_t)row * N + i] = (_Float16)v;
  }
}

// out = LN(a + b) * w + bias, fp32 + fp16 copies (N = 512; in-place safe on a)
__global__ __launch_bounds__(32)
void add_ln_kernel(const float* __restrict__ A, const float* __restrict__ Bv,
                   const float* __restrict__ w, const float* __restrict__ bvec,
                   float* __restrict__ out, _Float16* __restrict__ outh) {
  const int row = blockIdx.x, lane = threadIdx.x;
  const float* pa = A  + (size_t)row * D_SZ;
  const float* pb = Bv + (size_t)row * D_SZ;
  float vals[16];
  float s = 0.0f, s2 = 0.0f;
#pragma unroll
  for (int j = 0; j < 16; ++j) {
    const int i = lane + 32 * j;
    const float v = pa[i] + pb[i];
    vals[j] = v; s += v; s2 += v * v;
  }
#pragma unroll
  for (int off = 16; off > 0; off >>= 1) {
    s  += __shfl_xor(s,  off, 32);
    s2 += __shfl_xor(s2, off, 32);
  }
  const float mean = s / (float)D_SZ;
  const float var  = s2 / (float)D_SZ - mean * mean;
  const float rs   = rsqrtf(var + EPSF);
  float* q = out + (size_t)row * D_SZ;
  _Float16* qh = outh + (size_t)row * D_SZ;
#pragma unroll
  for (int j = 0; j < 16; ++j) {
    const int i = lane + 32 * j;
    const float v = (vals[j] - mean) * rs * w[i] + bvec[i];
    q[i] = v;
    qh[i] = (_Float16)v;
  }
}

// outh = fp16( feat0 + inorm(feat) )   (N = 512, feeds the MLP GEMM)
__global__ __launch_bounds__(32)
void inorm_add_kernel(const float* __restrict__ f0, const float* __restrict__ f,
                      _Float16* __restrict__ outh) {
  const int row = blockIdx.x, lane = threadIdx.x;
  const float* p = f + (size_t)row * D_SZ;
  float vals[16];
  float s = 0.0f, s2 = 0.0f;
#pragma unroll
  for (int j = 0; j < 16; ++j) {
    const int i = lane + 32 * j;
    const float v = p[i];
    vals[j] = v; s += v; s2 += v * v;
  }
#pragma unroll
  for (int off = 16; off > 0; off >>= 1) {
    s  += __shfl_xor(s,  off, 32);
    s2 += __shfl_xor(s2, off, 32);
  }
  const float mean = s / (float)D_SZ;
  const float var  = s2 / (float)D_SZ - mean * mean;
  const float rs   = rsqrtf(var + EPSF);
  _Float16* q = outh + (size_t)row * D_SZ;
  const float* q0 = f0 + (size_t)row * D_SZ;
#pragma unroll
  for (int j = 0; j < 16; ++j) {
    const int i = lane + 32 * j;
    q[i] = (_Float16)(q0[i] + (vals[j] - mean) * rs);
  }
}

// logits[row] = { -min_p cdist(feat,negatives), -min_p cdist(feat,highlights) }
__global__ __launch_bounds__(32)
void proto_kernel(const float* __restrict__ feat, const float* __restrict__ Hp,
                  const float* __restrict__ Np, float* __restrict__ out) {
  const int row = blockIdx.x, lane = threadIdx.x;
  float f[4];
#pragma unroll
  for (int j = 0; j < 4; ++j) f[j] = feat[(size_t)row * Q_SZ + lane * 4 + j];
  float bp = 3.4e38f, bn = 3.4e38f;
  for (int p = 0; p < P_SZ; ++p) {
    float d2 = 0.0f, e2 = 0.0f;
#pragma unroll
    for (int j = 0; j < 4; ++j) {
      const float dh = f[j] - Hp[p * Q_SZ + lane * 4 + j];
      const float dn = f[j] - Np[p * Q_SZ + lane * 4 + j];
      d2 += dh * dh;
      e2 += dn * dn;
    }
#pragma unroll
    for (int off = 16; off > 0; off >>= 1) {
      d2 += __shfl_xor(d2, off, 32);
      e2 += __shfl_xor(e2, off, 32);
    }
    bp = fminf(bp, d2);
    bn = fminf(bn, e2);
  }
  if (lane == 0) {
    out[row * 2 + 0] = -sqrtf(fmaxf(bn, 0.0f));
    out[row * 2 + 1] = -sqrtf(fmaxf(bp, 0.0f));
  }
}

// ---------------------------------------------------------------------------
extern "C" void kernel_launch(void* const* d_in, const int* in_sizes, int n_in,
                              void* d_out, int out_size, void* d_ws, size_t ws_size,
                              hipStream_t stream) {
  (void)in_sizes; (void)n_in; (void)out_size; (void)ws_size;
  const float* x        = (const float*)d_in[0];
  const float* reduce_w = (const float*)d_in[1];
  const float* reduce_b = (const float*)d_in[2];
  const float* in_w     = (const float*)d_in[3];
  const float* in_b     = (const float*)d_in[4];
  const float* out_w    = (const float*)d_in[5];
  const float* out_b    = (const float*)d_in[6];
  const float* ff1_w    = (const float*)d_in[7];
  const float* ff1_b    = (const float*)d_in[8];
  const float* ff2_w    = (const float*)d_in[9];
  const float* ff2_b    = (const float*)d_in[10];
  const float* ln1_w    = (const float*)d_in[11];
  const float* ln1_b    = (const float*)d_in[12];
  const float* ln2_w    = (const float*)d_in[13];
  const float* ln2_b    = (const float*)d_in[14];
  const float* mlp_w1   = (const float*)d_in[15];
  const float* mlp_b1   = (const float*)d_in[16];
  const float* mlp_w2   = (const float*)d_in[17];
  const float* mlp_b2   = (const float*)d_in[18];
  const float* mlp_w3   = (const float*)d_in[19];
  const float* mlp_b3   = (const float*)d_in[20];
  const float* highlights = (const float*)d_in[21];
  const float* negatives  = (const float*)d_in[22];

  // ---- fp32 scratch ----
  float* wsf   = (float*)d_ws;
  float* feat0 = wsf;                            // 2048*512
  float* feat  = feat0 + (size_t)B_SZ * D_SZ;
  float* tmp1  = feat  + (size_t)B_SZ * D_SZ;
  float* tmp2  = tmp1  + (size_t)B_SZ * D_SZ;
  float* featq = tmp2  + (size_t)B_SZ * D_SZ;    // 2048*128
  // ---- fp16 scratch ----
  _Float16* wsh     = (_Float16*)(featq + (size_t)B_SZ * Q_SZ);
  _Float16* xh      = wsh;                                   // 2048*768
  _Float16* wh_red  = xh      + (size_t)B_SZ * 768;          // 512*768
  _Float16* wh_in   = wh_red  + (size_t)D_SZ * 768;          // 3*1536*512
  _Float16* wh_out  = wh_in   + (size_t)3 * QKV_LD * D_SZ;   // 3*512*512
  _Float16* wh_ff1  = wh_out  + (size_t)3 * D_SZ * D_SZ;
  _Float16* wh_ff2  = wh_ff1  + (size_t)3 * D_SZ * D_SZ;
  _Float16* wh_m1   = wh_ff2  + (size_t)3 * D_SZ * D_SZ;     // 512*512
  _Float16* wh_m2   = wh_m1   + (size_t)D_SZ * D_SZ;
  _Float16* wh_m3   = wh_m2   + (size_t)D_SZ * D_SZ;         // 128*512
  _Float16* feat0h  = wh_m3   + (size_t)Q_SZ * D_SZ;         // 2048*512
  _Float16* feath   = feat0h  + (size_t)B_SZ * D_SZ;
  _Float16* qkvh    = feath   + (size_t)B_SZ * D_SZ;         // 2048*1536
  _Float16* ctxh    = qkvh    + (size_t)B_SZ * QKV_LD;       // 2048*512
  _Float16* t1h     = ctxh    + (size_t)B_SZ * D_SZ;
  _Float16* t2h     = t1h     + (size_t)B_SZ * D_SZ;

  const dim3 blk(32);
  const int MT64 = B_SZ / 64;  // 32 row tiles of 64

  // ---- one-time fp16 conversion of x and all GEMM weights ----
  struct CvtJob { const float* s; _Float16* d; int n; };
  const CvtJob jobs[] = {
    { x,        xh,     B_SZ * 768 },
    { reduce_w, wh_red, D_SZ * 768 },
    { in_w,     wh_in,  3 * QKV_LD * D_SZ },
    { out_w,    wh_out, 3 * D_SZ * D_SZ },
    { ff1_w,    wh_ff1, 3 * D_SZ * D_SZ },
    { ff2_w,    wh_ff2, 3 * D_SZ * D_SZ },
    { mlp_w1,   wh_m1,  D_SZ * D_SZ },
    { mlp_w2,   wh_m2,  D_SZ * D_SZ },
    { mlp_w3,   wh_m3,  Q_SZ * D_SZ },
  };
  for (const CvtJob& j : jobs)
    cvt_f32_f16_kernel<<<(j.n + 1023) / 1024, 256, 0, stream>>>(j.s, j.d, j.n);

  // feat0 = inorm(x @ reduce_w^T + reduce_b)   (fp32 + fp16 copies)
  gemm_bias_kernel<<<dim3(D_SZ / 64, MT64), blk, 0, stream>>>(
      xh, wh_red, reduce_b, tmp1, nullptr, D_SZ, 768, 0);
  inorm_kernel<<<B_SZ, blk, 0, stream>>>(tmp1, feat0, feat0h, D_SZ);

  const float*    curf = feat0;
  const _Float16* curh = feat0h;
  for (int i = 0; i < 3; ++i) {
    gemm_bias_kernel<<<dim3(QKV_LD / 64, MT64), blk, 0, stream>>>(
        curh, wh_in + (size_t)i * QKV_LD * D_SZ, in_b + (size_t)i * QKV_LD,
        nullptr, qkvh, QKV_LD, D_SZ, 0);
    flash_attn_kernel<<<dim3(B_SZ / 32, NH), blk, 0, stream>>>(qkvh, ctxh);
    gemm_bias_kernel<<<dim3(D_SZ / 64, MT64), blk, 0, stream>>>(
        ctxh, wh_out + (size_t)i * D_SZ * D_SZ, out_b + (size_t)i * D_SZ,
        tmp1, nullptr, D_SZ, D_SZ, 0);
    add_ln_kernel<<<B_SZ, blk, 0, stream>>>(
        curf, tmp1, ln1_w + (size_t)i * D_SZ, ln1_b + (size_t)i * D_SZ,
        feat, feath);
    gemm_bias_kernel<<<dim3(D_SZ / 64, MT64), blk, 0, stream>>>(
        feath, wh_ff1 + (size_t)i * D_SZ * D_SZ, ff1_b + (size_t)i * D_SZ,
        nullptr, t1h, D_SZ, D_SZ, 1);
    gemm_bias_kernel<<<dim3(D_SZ / 64, MT64), blk, 0, stream>>>(
        t1h, wh_ff2 + (size_t)i * D_SZ * D_SZ, ff2_b + (size_t)i * D_SZ,
        tmp2, nullptr, D_SZ, D_SZ, 0);
    add_ln_kernel<<<B_SZ, blk, 0, stream>>>(
        feat, tmp2, ln2_w + (size_t)i * D_SZ, ln2_b + (size_t)i * D_SZ,
        feat, feath);
    curf = feat;
    curh = feath;
  }

  // t1h = fp16( feat0 + inorm(feat) )
  inorm_add_kernel<<<B_SZ, blk, 0, stream>>>(feat0, feat, t1h);

  // MLP 512 -> 512 -> 512 -> 128
  gemm_bias_kernel<<<dim3(D_SZ / 64, MT64), blk, 0, stream>>>(
      t1h, wh_m1, mlp_b1, nullptr, t2h, D_SZ, D_SZ, 1);
  gemm_bias_kernel<<<dim3(D_SZ / 64, MT64), blk, 0, stream>>>(
      t2h, wh_m2, mlp_b2, nullptr, t1h, D_SZ, D_SZ, 1);
  gemm_bias_kernel<<<dim3(Q_SZ / 64, MT64), blk, 0, stream>>>(
      t1h, wh_m3, mlp_b3, featq, nullptr, Q_SZ, D_SZ, 0);
  inorm_kernel<<<B_SZ, blk, 0, stream>>>(featq, featq, nullptr, Q_SZ);

  proto_kernel<<<B_SZ, blk, 0, stream>>>(featq, highlights, negatives,
                                         (float*)d_out);
}
